// SparseNet_66872640799365
// MI455X (gfx1250) — matrix operand order repaired and checked by reading, not compiled, not run on hardware
//
#include <hip/hip_runtime.h>

typedef __bf16  bf16x16 __attribute__((ext_vector_type(16)));
typedef float   f32x8   __attribute__((ext_vector_type(8)));
typedef unsigned int u32x8 __attribute__((ext_vector_type(8)));

#define N_CH   256
#define MM     64
#define KK     9
#define COUT   72            // (M-1)*S + K
#define TAPS   81
#define TAPS_P 96            // padded to 3 k-steps of 32
#define BATCH  32
#define MSQ    4096          // M*M
#define CSQ    5184          // COUT*COUT
#define NPIX   (BATCH*MM*MM) // 131072 channel-last pixels
#define LMDA   0.005f
#define R_LR   1e-4f
#define ITERS  50
#define NBLK_GRAD (BATCH*MM) // 2048 partial-pair blocks

// ---------------- WMMA fragment helpers (wave32, 16x16x32 bf16) ----------------
// A (16x32, 16-bit): lane -> M row; VGPR v -> K pair (v>>2)*16 + (lane>>4)*8 + (v&3)*2
__device__ __forceinline__ bf16x16 frag_a_bf(const __bf16* base, int rstride, int lane) {
    int row = lane & 15, half = lane >> 4;
    const __bf16* p = base + row * rstride;
    u32x8 u;
#pragma unroll
    for (int v = 0; v < 8; ++v) {
        int k0 = (v >> 2) * 16 + half * 8 + (v & 3) * 2;
        u[v] = *(const unsigned int*)(p + k0);   // two consecutive bf16, 4B aligned
    }
    return __builtin_bit_cast(bf16x16, u);
}
// A from f32 source (convert to bf16 on the fly)
__device__ __forceinline__ bf16x16 frag_a_f32(const float* base, int rstride, int lane) {
    int row = lane & 15, half = lane >> 4;
    const float* p = base + row * rstride;
    bf16x16 a;
#pragma unroll
    for (int v = 0; v < 8; ++v) {
        int k0 = (v >> 2) * 16 + half * 8 + (v & 3) * 2;
        float2 f = *(const float2*)(p + k0);
        a[2 * v]     = (__bf16)f.x;
        a[2 * v + 1] = (__bf16)f.y;
    }
    return a;
}
// B (32x16, 16-bit): lane -> N col; VGPR v -> K pair (v>>2)*16 + (lane>>4)*8 + (v&3)*2 (same as A).
// Source layout: column-contiguous (element (k, col) at base[col*cstride + k]).
__device__ __forceinline__ bf16x16 frag_b_bf(const __bf16* base, int cstride, int lane) {
    int col = lane & 15, half = lane >> 4;
    const __bf16* p = base + col * cstride;
    u32x8 u;
#pragma unroll
    for (int v = 0; v < 8; ++v) {
        int k0 = (v >> 2) * 16 + half * 8 + (v & 3) * 2;
        u[v] = *(const unsigned int*)(p + k0);
    }
    return __builtin_bit_cast(bf16x16, u);
}
__device__ __forceinline__ f32x8 wmma_bf(bf16x16 a, bf16x16 b, f32x8 c) {
    return __builtin_amdgcn_wmma_f32_16x16x32_bf16(false, a, false, b, (short)0, c, false, false);
}

// ---------------- init kernels ----------------
__global__ void k_init_zero(float* R, int* flag) {
    long i = (long)blockIdx.x * blockDim.x + threadIdx.x;
    long stride = (long)gridDim.x * blockDim.x;
    for (long j = i; j < (long)NPIX * N_CH; j += stride) R[j] = 0.f;
    if (blockIdx.x == 0 && threadIdx.x == 0) *flag = 0;
}
__global__ void k_cvt_u(const float* U, __bf16* Ubf, __bf16* UbfT) {
    long i = (long)blockIdx.x * blockDim.x + threadIdx.x;
    long stride = (long)gridDim.x * blockDim.x;
    for (long j = i; j < (long)MSQ * CSQ; j += stride) {
        int m = (int)(j / CSQ), k = (int)(j - (long)m * CSQ);
        float f = U[j];
        Ubf[j] = (__bf16)f;                    // (m, k) row-major
        UbfT[(long)k * MSQ + m] = (__bf16)f;   // (k, m) row-major
    }
}
__global__ void k_cvt_w(const float* cw, __bf16* wCh, __bf16* wTp) {
    int i = blockIdx.x * blockDim.x + threadIdx.x;
    if (i >= N_CH * TAPS_P) return;
    int n = i / TAPS_P, t = i - n * TAPS_P;
    float f = (t < TAPS) ? cw[n * TAPS + t] : 0.f;
    wCh[n * TAPS_P + t] = (__bf16)f;   // (n, t): B for grad-conv GEMM
    wTp[t * N_CH + n]   = (__bf16)f;   // (t, n): B for conv_t GEMM
}

// ---------------- conv_t stage 1: s[pix, tap] = sum_n Rcl[pix, n] * w[n, tap] ----------------
// One wave = 16 pixels x all 96 taps: A fragment (the big R stream) loaded once per
// k-step and reused for 6 WMMAs against the 48KB L0-resident weight matrix.
__global__ void k_convt_gemm(const float* Rcl, const __bf16* wTp, float* s,
                             const int* flag, int check) {
    if (check && *flag) return;
    int wave = threadIdx.x >> 5, lane = threadIdx.x & 31;
    int pixtile = blockIdx.x * 4 + wave;   // 8192 tiles of 16 pixels
    const float* abase = Rcl + (long)pixtile * 16 * N_CH;
    f32x8 acc[6] = {};
#pragma unroll
    for (int ks = 0; ks < N_CH / 32; ++ks) {
        bf16x16 a = frag_a_f32(abase + ks * 32, N_CH, lane);
#pragma unroll
        for (int tt = 0; tt < 6; ++tt) {
            bf16x16 b = frag_b_bf(wTp + tt * 16 * N_CH + ks * 32, N_CH, lane);
            acc[tt] = wmma_bf(a, b, acc[tt]);
        }
    }
    int col = lane & 15, r0 = (lane >> 4) * 8;
#pragma unroll
    for (int tt = 0; tt < 6; ++tt)
#pragma unroll
        for (int v = 0; v < 8; ++v)
            s[(long)(pixtile * 16 + r0 + v) * TAPS_P + tt * 16 + col] = acc[tt][v];
}

// ---------------- conv_t stage 2: gather + bias -> c (bf16) ----------------
__global__ void k_convt_gather(const float* s, const float* bias, __bf16* cbf,
                               const int* flag, int check) {
    if (check && *flag) return;
    int i = blockIdx.x * blockDim.x + threadIdx.x;
    if (i >= BATCH * CSQ) return;
    int b = i / CSQ, r = i - b * CSQ;
    int oy = r / COUT, ox = r - oy * COUT;
    float acc = bias[0];
    int ky0 = max(0, oy - (MM - 1)), ky1 = min(KK - 1, oy);
    int kx0 = max(0, ox - (MM - 1)), kx1 = min(KK - 1, ox);
    for (int ky = ky0; ky <= ky1; ++ky)
        for (int kx = kx0; kx <= kx1; ++kx)
            acc += s[(long)(((b * MM) + (oy - ky)) * MM + (ox - kx)) * TAPS_P + ky * KK + kx];
    cbf[i] = (__bf16)acc;
}

// ---------------- GEMM1: pred = c @ U^T; emit g2 = bf16(-2*(img - pred)) or final pred ----------------
// 2-way N register blocking: one A fragment drives 2 WMMAs; prefetch U stream ahead.
__global__ void k_gemm1(const __bf16* cbf, const __bf16* Ubf, const float* img,
                        __bf16* g2, float* outp, const int* flag, int check, int fin) {
    if (check && *flag) return;
    int wave = threadIdx.x >> 5, lane = threadIdx.x & 31;
    int btile = wave;                 // 2 tiles of 16 batch rows
    int mt = blockIdx.x * 2;          // 128 blocks x 2 col tiles
    const __bf16* abase = cbf + btile * 16 * CSQ;
    const __bf16* bb0 = Ubf + (long)mt * 16 * CSQ;
    const __bf16* bb1 = bb0 + (long)16 * CSQ;
    int prow = (lane & 15) * CSQ;
    f32x8 acc0 = {}, acc1 = {};
#pragma unroll 2
    for (int ks = 0; ks < CSQ / 32; ++ks) {
        if ((ks & 7) == 0) {          // prefetch 8 k-steps (512B) ahead per B row
            __builtin_prefetch(bb0 + prow + ks * 32 + 256, 0, 1);
            __builtin_prefetch(bb1 + prow + ks * 32 + 256, 0, 1);
        }
        bf16x16 a  = frag_a_bf(abase + ks * 32, CSQ, lane);
        bf16x16 f0 = frag_b_bf(bb0 + ks * 32, CSQ, lane);
        bf16x16 f1 = frag_b_bf(bb1 + ks * 32, CSQ, lane);
        acc0 = wmma_bf(a, f0, acc0);
        acc1 = wmma_bf(a, f1, acc1);
    }
    int col = lane & 15, r0 = (lane >> 4) * 8;
#pragma unroll
    for (int v = 0; v < 8; ++v) {
        int brow = btile * 16 + r0 + v;
        int m0 = mt * 16 + col, m1 = m0 + 16;
        if (fin) {
            outp[brow * MSQ + m0] = acc0[v];
            outp[brow * MSQ + m1] = acc1[v];
        } else {
            g2[brow * MSQ + m0] = (__bf16)(-2.f * (img[brow * MSQ + m0] - acc0[v]));
            g2[brow * MSQ + m1] = (__bf16)(-2.f * (img[brow * MSQ + m1] - acc1[v]));
        }
    }
}

// ---------------- GEMM2: dc = g2 @ U (via U^T rows), 2-way N blocking ----------------
__global__ void k_gemm2(const __bf16* g2, const __bf16* UbfT, __bf16* dcb, const int* flag) {
    if (*flag) return;
    int wave = threadIdx.x >> 5, lane = threadIdx.x & 31;
    int btile = wave;                 // 2
    int kt = blockIdx.x * 2;          // 162 blocks x 2 col tiles
    const __bf16* abase = g2 + btile * 16 * MSQ;
    const __bf16* bb0 = UbfT + (long)kt * 16 * MSQ;
    const __bf16* bb1 = bb0 + (long)16 * MSQ;
    int prow = (lane & 15) * MSQ;
    f32x8 acc0 = {}, acc1 = {};
#pragma unroll 2
    for (int ks = 0; ks < MSQ / 32; ++ks) {
        if ((ks & 7) == 0) {
            __builtin_prefetch(bb0 + prow + ks * 32 + 256, 0, 1);
            __builtin_prefetch(bb1 + prow + ks * 32 + 256, 0, 1);
        }
        bf16x16 a  = frag_a_bf(abase + ks * 32, MSQ, lane);
        bf16x16 f0 = frag_b_bf(bb0 + ks * 32, MSQ, lane);
        bf16x16 f1 = frag_b_bf(bb1 + ks * 32, MSQ, lane);
        acc0 = wmma_bf(a, f0, acc0);
        acc1 = wmma_bf(a, f1, acc1);
    }
    int col = lane & 15, r0 = (lane >> 4) * 8;
#pragma unroll
    for (int v = 0; v < 8; ++v) {
        int brow = btile * 16 + r0 + v;
        dcb[brow * CSQ + kt * 16 + col]      = acc0[v];
        dcb[brow * CSQ + kt * 16 + 16 + col] = acc1[v];
    }
}

// ---------------- grad conv (implicit GEMM) + ISTA update + norm partials ----------------
// Block = (b, iy), 4 waves; each wave builds the im2col A fragments ONCE per ix-tile
// and reuses them across its 4 channel tiles (B is 48KB, L0-resident).
__global__ void k_grad_update(const __bf16* dcb, const __bf16* wCh, float* Rcl,
                              float* partials, const int* flag) {
    if (*flag) return;
    int wave = threadIdx.x >> 5, lane = threadIdx.x & 31;
    int b = blockIdx.x, iy = blockIdx.y;
    const __bf16* dcrow = dcb + b * CSQ;
    int row = lane & 15, half = lane >> 4;
    float d2 = 0.f, r2 = 0.f;
    for (int ix0 = 0; ix0 < MM; ix0 += 16) {
        bf16x16 a[3];
#pragma unroll
        for (int ks = 0; ks < 3; ++ks) {
#pragma unroll
            for (int v = 0; v < 8; ++v) {
                int k0 = (v >> 2) * 16 + half * 8 + (v & 3) * 2;
#pragma unroll
                for (int e = 0; e < 2; ++e) {
                    int t = ks * 32 + k0 + e;
                    __bf16 val = (__bf16)0.f;
                    if (t < TAPS) {
                        int ky = t / KK, kx = t - ky * KK;
                        val = dcrow[(iy + ky) * COUT + ix0 + row + kx];
                    }
                    a[ks][2 * v + e] = val;
                }
            }
        }
#pragma unroll
        for (int j = 0; j < 4; ++j) {
            int ntile = wave * 4 + j;
            f32x8 acc = {};
#pragma unroll
            for (int ks = 0; ks < 3; ++ks) {
                bf16x16 bw = frag_b_bf(wCh + ntile * 16 * TAPS_P + ks * 32, TAPS_P, lane);
                acc = wmma_bf(a[ks], bw, acc);
            }
            int col = lane & 15, r0 = (lane >> 4) * 8;
            int nch = ntile * 16 + col;
#pragma unroll
            for (int v = 0; v < 8; ++v) {
                int ix = ix0 + r0 + v;
                long idx = (((long)(b * MM + iy)) * MM + ix) * N_CH + nch;
                float old = Rcl[idx];
                float x = old - R_LR * acc[v];
                float ax = fabsf(x) - LMDA;
                float rn = ax > 0.f ? copysignf(ax, x) : 0.f;
                Rcl[idx] = rn;
                float d = rn - old;
                d2 += d * d;
                r2 += old * old;
            }
        }
    }
    __shared__ float sd[128], sr[128];
    sd[threadIdx.x] = d2; sr[threadIdx.x] = r2;
    __syncthreads();
    for (int st = 64; st > 0; st >>= 1) {
        if (threadIdx.x < st) { sd[threadIdx.x] += sd[threadIdx.x + st]; sr[threadIdx.x] += sr[threadIdx.x + st]; }
        __syncthreads();
    }
    if (threadIdx.x == 0) {
        int bid = blockIdx.x + gridDim.x * blockIdx.y;   // 2048 blocks
        partials[2 * bid] = sd[0];
        partials[2 * bid + 1] = sr[0];
    }
}

// ---------------- deterministic convergence reduction ----------------
__global__ void k_reduce(const float* partials, int* flag) {
    if (*flag) return;
    __shared__ float sn[256], sdn[256];
    float num = 0.f, den = 0.f;
    for (int i = threadIdx.x; i < NBLK_GRAD; i += 256) {
        num += partials[2 * i];
        den += partials[2 * i + 1];
    }
    sn[threadIdx.x] = num; sdn[threadIdx.x] = den;
    __syncthreads();
    for (int st = 128; st > 0; st >>= 1) {
        if (threadIdx.x < st) { sn[threadIdx.x] += sn[threadIdx.x + st]; sdn[threadIdx.x] += sdn[threadIdx.x + st]; }
        __syncthreads();
    }
    // change = sqrt(num)/sqrt(den) < 0.01  <=>  num < 1e-4 * den  (den=0 -> keep going)
    if (threadIdx.x == 0 && sn[0] < 1e-4f * sdn[0]) *flag = 1;
}

extern "C" void kernel_launch(void* const* d_in, const int* in_sizes, int n_in,
                              void* d_out, int out_size, void* d_ws, size_t ws_size,
                              hipStream_t stream) {
    const float* img   = (const float*)d_in[0];
    const float* U     = (const float*)d_in[1];
    const float* convw = (const float*)d_in[2];
    const float* convb = (const float*)d_in[3];

    char* w = (char*)d_ws;
    size_t off = 0;
    auto alloc = [&](size_t bytes) -> char* {
        char* p = w + off;
        off = (off + bytes + 255) & ~(size_t)255;
        return p;
    };
    float*  R    = (float*) alloc((size_t)NPIX * N_CH * 4);     // 134 MB, channel-last fp32
    float*  s    = (float*) alloc((size_t)NPIX * TAPS_P * 4);   // 50 MB
    __bf16* Ubf  = (__bf16*)alloc((size_t)MSQ * CSQ * 2);       // 42 MB
    __bf16* UbfT = (__bf16*)alloc((size_t)MSQ * CSQ * 2);       // 42 MB
    __bf16* cbf  = (__bf16*)alloc((size_t)BATCH * CSQ * 2);
    __bf16* g2   = (__bf16*)alloc((size_t)BATCH * MSQ * 2);
    __bf16* dcb  = (__bf16*)alloc((size_t)BATCH * CSQ * 2);
    __bf16* wCh  = (__bf16*)alloc((size_t)N_CH * TAPS_P * 2);
    __bf16* wTp  = (__bf16*)alloc((size_t)N_CH * TAPS_P * 2);
    float*  part = (float*) alloc((size_t)NBLK_GRAD * 2 * 4);
    int*    flag = (int*)   alloc(256);
    float*  outp = (float*)d_out;

    k_init_zero<<<4096, 256, 0, stream>>>(R, flag);
    k_cvt_u<<<4096, 256, 0, stream>>>(U, Ubf, UbfT);
    k_cvt_w<<<(N_CH * TAPS_P + 255) / 256, 256, 0, stream>>>(convw, wCh, wTp);

    for (int it = 0; it < ITERS; ++it) {
        k_convt_gemm<<<NPIX / 64, 128, 0, stream>>>(R, wTp, s, flag, 1);
        k_convt_gather<<<(BATCH * CSQ + 255) / 256, 256, 0, stream>>>(s, convb, cbf, flag, 1);
        k_gemm1<<<MSQ / 32, 64, 0, stream>>>(cbf, Ubf, img, g2, outp, flag, 1, 0);
        k_gemm2<<<CSQ / 32, 64, 0, stream>>>(g2, UbfT, dcb, flag);
        k_grad_update<<<dim3(BATCH, MM), 128, 0, stream>>>(dcb, wCh, R, part, flag);
        k_reduce<<<1, 256, 0, stream>>>(part, flag);
    }
    // final: c = conv_t(R_final); out = c @ U^T  (run unconditionally)
    k_convt_gemm<<<NPIX / 64, 128, 0, stream>>>(R, wTp, s, flag, 0);
    k_convt_gather<<<(BATCH * CSQ + 255) / 256, 256, 0, stream>>>(s, convb, cbf, flag, 0);
    k_gemm1<<<MSQ / 32, 64, 0, stream>>>(cbf, Ubf, img, g2, outp, flag, 0, 1);
}
